// LSTM_86921548136836
// MI455X (gfx1250) — matrix operand order repaired and checked by reading, not compiled, not run on hardware
//
#include <hip/hip_runtime.h>
#include <hip/hip_bf16.h>

typedef __bf16 bf16;
typedef __attribute__((ext_vector_type(16))) __bf16 v16bf;
typedef __attribute__((ext_vector_type(8)))  __bf16 v8bf;
typedef __attribute__((ext_vector_type(8)))  float   v8f;
typedef __attribute__((ext_vector_type(4)))  float   v4f;

#define Tn 2048
#define Bn 64
#define Fn 512
#define Hn 512
#define C4 2048                     // 4*H gate columns
#define MROWS (Bn * Tn)             // 131072 (b,t) rows

// ---- workspace layout (bytes). Requires ~1.13 GiB of d_ws. ----
#define OFF_XP   ((size_t)0)                                   // Xp fp32 [T][4H][B]  = 1 GiB
#define OFF_XBF  (OFF_XP   + (size_t)Tn * C4 * Bn * 4)          // x bf16 [B*T][F]    = 128 MiB
#define OFF_WTI  (OFF_XBF  + (size_t)MROWS * Fn * 2)            // W_i^T bf16 [4H][F] = 2 MiB
#define OFF_WTH  (OFF_WTI  + (size_t)C4 * Fn * 2)               // W_h^T bf16 [4H][H] = 2 MiB
#define OFF_HBUF (OFF_WTH  + (size_t)C4 * Fn * 2)               // h ping-pong bf16 [2][B][H]
#define OFF_BSUM (OFF_HBUF + (size_t)2 * Bn * Hn * 2)           // bias_i+bias_h fp32 [4H]
#define OFF_SYNC (OFF_BSUM + (size_t)C4 * 4)                    // barrier words

#define KP   520     // padded LDS K-stride (bf16 elems): 1040 B rows -> conflict-free b128
#define NBLK 8       // persistent workgroups in recurrent phase (64 cols each)

// ---------------- fragment loaders (bf16 WMMA, wave32) ----------------
// A 16x32: lane L (m = L&15, half = L>>4): elems 0..7 = K(half*8..+7), 8..15 = K(16+half*8..+7)
__device__ inline v16bf load_a_frag(const bf16* p /* rowbase + k0 + half*8 */) {
    v16bf a;
    *(v8bf*)&a       = *(const v8bf*)p;
    *((v8bf*)&a + 1) = *(const v8bf*)(p + 16);
    return a;
}
// B 32x16 (from transposed/column-major storage): lane L (n = L&15, half = L>>4):
// elems 0..7 = K(half*8..+7), 8..15 = K(16+half*8..+7) of column n
__device__ inline v16bf load_b_frag(const bf16* p /* colbase + k0 + half*8 */) {
    v16bf b;
    *(v8bf*)&b       = *(const v8bf*)p;
    *((v8bf*)&b + 1) = *(const v8bf*)(p + 16);
    return b;
}

__device__ inline float sigmoidf_(float x) { return 1.0f / (1.0f + __expf(-x)); }

// ---------------- kernel 0: convert / transpose / init ----------------
__global__ void lstm_prep_kernel(
    const float* __restrict__ x,
    const float* Wii, const float* Whi, const float* Wif, const float* Whf,
    const float* Wig, const float* Whg, const float* Wio, const float* Who,
    const float* bii, const float* bhi, const float* bif, const float* bhf,
    const float* big, const float* bhg, const float* bio, const float* bho,
    char* __restrict__ ws)
{
    bf16*  xbf  = (bf16*)(ws + OFF_XBF);
    bf16*  wti  = (bf16*)(ws + OFF_WTI);
    bf16*  wth  = (bf16*)(ws + OFF_WTH);
    bf16*  hbuf = (bf16*)(ws + OFF_HBUF);
    float* bsum = (float*)(ws + OFF_BSUM);
    unsigned* sy = (unsigned*)(ws + OFF_SYNC);

    const float* Wi[4] = {Wii, Wif, Wig, Wio};
    const float* Wh[4] = {Whi, Whf, Whg, Who};
    const float* bi[4] = {bii, bif, big, bio};
    const float* bh[4] = {bhi, bhf, bhg, bho};

    size_t tid = (size_t)blockIdx.x * blockDim.x + threadIdx.x;
    size_t nth = (size_t)gridDim.x * blockDim.x;

    for (size_t i = tid; i < (size_t)MROWS * Fn; i += nth)
        xbf[i] = (bf16)x[i];

    // W^T, bf16: wt[(g*512 + j)*512 + k] = W_g[k][j]
    for (size_t i = tid; i < (size_t)C4 * Fn; i += nth) {
        int g = (int)(i >> 18);
        int rem = (int)(i & 262143);
        int j = rem >> 9, k = rem & 511;
        wti[i] = (bf16)Wi[g][(size_t)k * Hn + j];
        wth[i] = (bf16)Wh[g][(size_t)k * Hn + j];
    }
    for (size_t i = tid; i < (size_t)C4; i += nth) {
        int g = (int)(i >> 9), j = (int)(i & 511);
        bsum[i] = bi[g][j] + bh[g][j];
    }
    for (size_t i = tid; i < (size_t)2 * Bn * Hn; i += nth)
        hbuf[i] = (bf16)0.0f;           // h0 = 0 (both ping-pong buffers)
    if (tid < 4) sy[tid] = 0u;          // barrier counter + generation
}

// ---------------- kernel 1: Xp[t][c][b] = x_t @ W_i + (b_i + b_h) ----------------
// One block per timestep t (blockIdx.y = t): tile = 64 b-rows x 128 cols.
// Wave: 16 b-rows x 64 cols (4 WMMA tiles, A-frag reuse). The C-fragment's v index
// then walks the contiguous b dimension of Xp[t][c][b] -> 32 B-sector stores.
__global__ __launch_bounds__(256) void lstm_xproj_kernel(char* __restrict__ ws)
{
    const bf16*  xbf  = (const bf16*)(ws + OFF_XBF);
    const bf16*  wti  = (const bf16*)(ws + OFF_WTI);
    const float* bsum = (const float*)(ws + OFF_BSUM);
    float*       Xp   = (float*)(ws + OFF_XP);

    int lane = threadIdx.x & 31;
    int w    = threadIdx.x >> 5;
    int mt   = w & 3, jh = w >> 2;
    int n    = lane & 15, half = lane >> 4;

    int t  = blockIdx.y;                   // timestep handled by this block
    int b0 = 16 * mt;                      // batch-row base for this wave
    int c0 = blockIdx.x * 128 + 64 * jh;   // col base for this wave

    v8f acc[4] = {};
    // A row for this lane: x[b0+n][t][:]  (x is [B][T][F] row-major)
    const bf16* arow = xbf + ((size_t)(b0 + n) * Tn + t) * Fn;

    for (int k0 = 0; k0 < Fn; k0 += 32) {
        v16bf a = load_a_frag(arow + k0 + half * 8);
#pragma unroll
        for (int jt = 0; jt < 4; ++jt) {
            const bf16* pb = wti + (size_t)(c0 + 16 * jt + n) * Fn + k0 + half * 8;
            v16bf b = load_b_frag(pb);
            acc[jt] = __builtin_amdgcn_wmma_f32_16x16x32_bf16(
                false, a, false, b, (short)0, acc[jt], false, false);
        }
    }

#pragma unroll
    for (int jt = 0; jt < 4; ++jt) {
        int c = c0 + 16 * jt + n;
        float bias = bsum[c];
        // C layout: m = v + 8*half -> b = b0 + 8*half + v  (contiguous in v)
        float* p = Xp + ((size_t)t * C4 + c) * Bn + b0 + 8 * half;
        v4f s0 = { acc[jt][0] + bias, acc[jt][1] + bias,
                   acc[jt][2] + bias, acc[jt][3] + bias };
        v4f s1 = { acc[jt][4] + bias, acc[jt][5] + bias,
                   acc[jt][6] + bias, acc[jt][7] + bias };
        *(v4f*)p       = s0;
        *(v4f*)(p + 4) = s1;
    }
}

// ---------------- device-wide barrier (NBLK resident workgroups) ----------------
__device__ inline void grid_sync(unsigned* cnt, unsigned* gen, unsigned nb, unsigned target) {
    __syncthreads();
    if (threadIdx.x == 0) {
        __threadfence();
        unsigned prev = __hip_atomic_fetch_add(cnt, 1u, __ATOMIC_ACQ_REL,
                                               __HIP_MEMORY_SCOPE_AGENT);
        if (prev == nb - 1u) {
            __hip_atomic_store(cnt, 0u, __ATOMIC_RELAXED, __HIP_MEMORY_SCOPE_AGENT);
            __hip_atomic_fetch_add(gen, 1u, __ATOMIC_ACQ_REL, __HIP_MEMORY_SCOPE_AGENT);
        } else {
            while (__hip_atomic_load(gen, __ATOMIC_ACQUIRE, __HIP_MEMORY_SCOPE_AGENT) < target) {}
        }
    }
    __syncthreads();
}

// ---------------- kernel 2: sequential recurrence, W_h pinned in LDS ----------------
// 8 persistent WGs x 256 thr. WG owns 64 gate-cols (same j-slice for all 4 gates):
// 4 x 64 x 512 bf16 weights = 256 KB (padded 260 KB) resident in the 320 KB WGP LDS.
// Wave: all 4 gates for a 16b x 32j patch -> c stays in registers, no gate exchange.
// Xp[t] loads hoisted above the K-loop (latency hidden behind 128 WMMAs/wave);
// Xp[t+1] prefetched into cache via global_prefetch each step.
__global__ __launch_bounds__(256) void lstm_seq_kernel(char* __restrict__ ws,
                                                       float* __restrict__ out)
{
    extern __shared__ char smem[];
    bf16* WhS = (bf16*)smem;

    const bf16*  wth  = (const bf16*)(ws + OFF_WTH);
    const float* Xp   = (const float*)(ws + OFF_XP);
    bf16*        hbuf = (bf16*)(ws + OFF_HBUF);
    unsigned*    sy   = (unsigned*)(ws + OFF_SYNC);

    int tid = threadIdx.x;
    int jc  = blockIdx.x * 64;

    // preload W_h^T slice into LDS with padded stride (8 B chunks)
    for (int idx = tid; idx < 4 * 64 * (Hn / 4); idx += 256) {
        int row = idx / (Hn / 4);
        int k4  = (idx % (Hn / 4)) * 4;
        int g = row >> 6, jj = row & 63;
        const bf16* src = wth + (size_t)(g * Hn + jc + jj) * Hn + k4;
        bf16*       dst = WhS + (size_t)row * KP + k4;
        *(uint2*)dst = *(const uint2*)src;
    }
    __syncthreads();

    int lane = threadIdx.x & 31;
    int w    = threadIdx.x >> 5;
    int mt   = w & 3, jh = w >> 2;           // wave: rows 16*mt, j-tiles {2*jh, 2*jh+1}
    int n    = lane & 15, half = lane >> 4;
    int brow = 16 * mt + 8 * half;           // C layout: m = v + 8*half -> b = brow + v

    float cst[2][8];
#pragma unroll
    for (int jl = 0; jl < 2; ++jl)
#pragma unroll
        for (int v = 0; v < 8; ++v) cst[jl][v] = 0.0f;

    for (int t = 0; t < Tn; ++t) {
        // ---- issue Xp[t] loads early: constant data, independent of h ----
        float xg_[2][4][8];
#pragma unroll
        for (int jl = 0; jl < 2; ++jl) {
            int jglob = jc + 16 * (2 * jh + jl) + n;
#pragma unroll
            for (int g = 0; g < 4; ++g) {
                const float* p = Xp + ((size_t)t * C4 + g * Hn + jglob) * (size_t)Bn + brow;
                v4f lo = *(const v4f*)p;
                v4f hi = *(const v4f*)(p + 4);
#pragma unroll
                for (int v = 0; v < 4; ++v) { xg_[jl][g][v] = lo[v]; xg_[jl][g][v + 4] = hi[v]; }
            }
        }
        // ---- prefetch Xp[t+1] toward cache (gfx1250 global_prefetch) ----
        if (t + 1 < Tn) {
#pragma unroll
            for (int jl = 0; jl < 2; ++jl) {
                int jglob = jc + 16 * (2 * jh + jl) + n;
#pragma unroll
                for (int g = 0; g < 4; ++g)
                    __builtin_prefetch(
                        Xp + ((size_t)(t + 1) * C4 + g * Hn + jglob) * (size_t)Bn + brow, 0, 1);
            }
        }

        const bf16* hread  = hbuf + (size_t)(t & 1) * Bn * Hn;
        bf16*       hwrite = hbuf + (size_t)((t + 1) & 1) * Bn * Hn;
        const bf16* arow   = hread + (size_t)(16 * mt + n) * Hn;

        v8f acc[4][2] = {};
        for (int k0 = 0; k0 < Hn; k0 += 32) {
            v16bf a = load_a_frag(arow + k0 + half * 8);
#pragma unroll
            for (int g = 0; g < 4; ++g)
#pragma unroll
                for (int jl = 0; jl < 2; ++jl) {
                    int jt = 2 * jh + jl;
                    const bf16* pb = WhS + (size_t)(g * 64 + 16 * jt + n) * KP
                                     + k0 + half * 8;
                    v16bf b = load_b_frag(pb);
                    acc[g][jl] = __builtin_amdgcn_wmma_f32_16x16x32_bf16(
                        false, a, false, b, (short)0, acc[g][jl], false, false);
                }
        }

#pragma unroll
        for (int jl = 0; jl < 2; ++jl) {
            int jglob = jc + 16 * (2 * jh + jl) + n;
#pragma unroll
            for (int v = 0; v < 8; ++v) {
                float xi = xg_[jl][0][v] + acc[0][jl][v];
                float xf = xg_[jl][1][v] + acc[1][jl][v];
                float xg = xg_[jl][2][v] + acc[2][jl][v];
                float xo = xg_[jl][3][v] + acc[3][jl][v];
                float ig = sigmoidf_(xi);
                float fg = sigmoidf_(xf);
                float gg = tanhf(xg);
                float og = sigmoidf_(xo);
                float cn = fg * cst[jl][v] + ig * gg;
                cst[jl][v] = cn;
                float hn = og * tanhf(cn);
                int b = brow + v;
                hwrite[(size_t)b * Hn + jglob] = (bf16)hn;
                if (t == Tn - 1) {
                    out[(size_t)b * Hn + jglob] = hn;                       // h_T
                    out[(size_t)Bn * Hn + (size_t)b * Hn + jglob] = cn;     // c_T
                }
            }
        }
        grid_sync(sy, sy + 1, NBLK, (unsigned)(t + 1));
    }
}

// ---------------- launch ----------------
extern "C" void kernel_launch(void* const* d_in, const int* in_sizes, int n_in,
                              void* d_out, int out_size, void* d_ws, size_t ws_size,
                              hipStream_t stream) {
    const float* x   = (const float*)d_in[0];
    const float* Wii = (const float*)d_in[1];
    const float* Whi = (const float*)d_in[2];
    const float* Wif = (const float*)d_in[3];
    const float* Whf = (const float*)d_in[4];
    const float* Wig = (const float*)d_in[5];
    const float* Whg = (const float*)d_in[6];
    const float* Wio = (const float*)d_in[7];
    const float* Who = (const float*)d_in[8];
    const float* bii = (const float*)d_in[9];
    const float* bhi = (const float*)d_in[10];
    const float* bif = (const float*)d_in[11];
    const float* bhf = (const float*)d_in[12];
    const float* big = (const float*)d_in[13];
    const float* bhg = (const float*)d_in[14];
    const float* bio = (const float*)d_in[15];
    const float* bho = (const float*)d_in[16];

    char*  ws  = (char*)d_ws;
    float* out = (float*)d_out;

    lstm_prep_kernel<<<4096, 256, 0, stream>>>(
        x, Wii, Whi, Wif, Whf, Wig, Whg, Wio, Who,
        bii, bhi, bif, bhf, big, bhg, bio, bho, ws);

    lstm_xproj_kernel<<<dim3(C4 / 128, Tn), 256, 0, stream>>>(ws);

    size_t smem = (size_t)4 * 64 * KP * sizeof(bf16);   // 266,240 B < 320 KB WGP LDS
    lstm_seq_kernel<<<NBLK, 256, smem, stream>>>(ws, out);
}
